// TemporalAttention_2156073583147
// MI455X (gfx1250) — matrix operand order repaired and checked by reading, not compiled, not run on hardware
//
#include <hip/hip_runtime.h>

// Problem constants (from reference)
constexpr int BB  = 16;    // batch
constexpr int LQ  = 1024;  // query length
constexpr int LK  = 4096;  // key length
constexpr int DD  = 512;   // model dim (QD == KD)

typedef __attribute__((ext_vector_type(16))) _Float16 v16h;
typedef __attribute__((ext_vector_type(8)))  float    v8f;

// ---------------------------------------------------------------------------
// Fragment load helpers (this code's contiguous-k layout, wave32)
// ---------------------------------------------------------------------------

// B fragment (16-bit, 32x16 KxN): lane n = lane&15, K contiguous starting at
// (lane>>4)*16. Caller passes pointer to the 16 contiguous halfs. 16B aligned.
static __device__ inline v16h ld_bfrag(const _Float16* p) {
  union { v16h h; uint4 u[2]; } t;
  t.u[0] = *(const uint4*)(p);
  t.u[1] = *(const uint4*)(p + 8);
  return t.h;
}

// A fragment (16-bit, 16x32 MxK) from f16 row: elements [kbase..kbase+15]
// of the lane's row (row pointer already offset to k0).
static __device__ inline v16h ld_afrag(const _Float16* rowk, int kbase) {
  union { v16h h; uint4 u[2]; } t;
  t.u[0] = *(const uint4*)(rowk + kbase);
  t.u[1] = *(const uint4*)(rowk + kbase + 8);
  return t.h;
}

// ---------------------------------------------------------------------------
// Kernel 1: weight convert + transpose:  WT[n][k] = (f16) W[k][n]
// ---------------------------------------------------------------------------
__global__ __launch_bounds__(256) void wt_kernel(const float* __restrict__ W,
                                                 _Float16* __restrict__ WT) {
  int idx = blockIdx.x * 256 + threadIdx.x;     // 0 .. 512*512-1
  int k = idx / DD;
  int n = idx % DD;                             // coalesced read of W[k][n]
  WT[(size_t)n * DD + k] = (_Float16)W[idx];
}

// ---------------------------------------------------------------------------
// Kernel 2: projection GEMM  out = (A[M,512] @ W[512,512] + bias) * scale
//   A: f32 or f16 (templated), staged once to LDS as f16.  W given
//   transposed f16: WT[n][k].  Output f16, either row-major [M][512] or
//   (TRANSV) vT layout [batch][512][LK]; TRANSV stores go through a
//   per-wave 16x16 LDS transpose so each lane emits one coalesced 16B store.
//   Block: 256 threads = 8 waves; tile = 16 rows x 512 cols, 64 cols/wave.
// ---------------------------------------------------------------------------
template <typename AT, bool TRANSV>
__global__ __launch_bounds__(256) void proj_kernel(const AT* __restrict__ A,
                                                   const _Float16* __restrict__ WT,
                                                   const float* __restrict__ bias,
                                                   _Float16* __restrict__ out,
                                                   float scale) {
  __shared__ __align__(16) _Float16 sA[16 * DD];     // 16 KB activation tile
  __shared__ __align__(16) _Float16 sT[8][16 * 16];  // per-wave transpose buf

  const int lane   = threadIdx.x & 31;
  const int wave   = threadIdx.x >> 5;          // 0..7
  const int lm     = lane & 15;
  const int lhalf  = lane >> 4;                 // 0/1
  const int row0   = blockIdx.x * 16;
  const int colbas = wave * 64;
  const int kbaseA = lhalf * 16;
  const int kbaseB = lhalf * 16;

  // Stage (and convert) the 16x512 A tile into LDS
  const AT* Ablk = A + (size_t)row0 * DD;
  for (int i = threadIdx.x; i < 16 * DD; i += 256) sA[i] = (_Float16)Ablk[i];
  __syncthreads();

  v8f acc[4] = {};
  for (int k0 = 0; k0 < DD; k0 += 32) {
    v16h a = ld_afrag(&sA[lm * DD + k0], kbaseA);
#pragma unroll
    for (int ct = 0; ct < 4; ++ct) {
      int n = colbas + ct * 16 + lm;
      v16h b = ld_bfrag(WT + (size_t)n * DD + k0 + kbaseB);
      acc[ct] = __builtin_amdgcn_wmma_f32_16x16x32_f16(
          false, a, false, b, (short)0, acc[ct], false, false);
    }
  }

  if (TRANSV) {
    // vT[b][n][j] with the tile's 16 j's contiguous: LDS transpose per wave,
    // then each lane stores 8 contiguous halfs (one b128).
    const int bidx = row0 >> 12;                // / LK (4096)
    const int jb   = row0 & (LK - 1);
#pragma unroll
    for (int ct = 0; ct < 4; ++ct) {
      int n = colbas + ct * 16 + lm;
      float bv = bias[n];
#pragma unroll
      for (int r = 0; r < 8; ++r)
        sT[wave][(r + lhalf * 8) * 16 + lm] =
            (_Float16)((acc[ct][r] + bv) * scale);
      // same-wave DS ordering: reads below see the writes above
      union { _Float16 h[8]; uint4 u; } tr;
#pragma unroll
      for (int jj = 0; jj < 8; ++jj)
        tr.h[jj] = sT[wave][(lhalf * 8 + jj) * 16 + lm];
      _Float16* dst = out + ((size_t)bidx * DD + n) * LK + jb + lhalf * 8;
      *(uint4*)dst = tr.u;
    }
  } else {
#pragma unroll
    for (int ct = 0; ct < 4; ++ct) {
      int n = colbas + ct * 16 + lm;
      float bv = bias[n];
#pragma unroll
      for (int r = 0; r < 8; ++r) {
        int mm = row0 + r + lhalf * 8;          // C/D row layout
        out[(size_t)mm * DD + n] = (_Float16)((acc[ct][r] + bv) * scale);
      }
    }
  }
}

// ---------------------------------------------------------------------------
// Kernel 3: flash attention.
//   Block = 256 threads (8 wave32s) handles 16 queries of one batch.
//   Loop over key tiles of 128 with only TWO barriers per tile (sP is
//   double-buffered so the context phase of tile t can overlap the score
//   phase of tile t+1 on other waves):
//     score phase:   wave w computes 16x16 scores for keys [j0+16w, +16)
//     barrier
//     softmax phase: wave w owns rows {2w, 2w+1}; wave32 __shfl_xor reduce
//     barrier
//     context phase: wave w accumulates d-slice [64w, 64w+64) via P @ vT
// ---------------------------------------------------------------------------
__global__ __launch_bounds__(256) void attn_kernel(const _Float16* __restrict__ qh,
                                                   const _Float16* __restrict__ kh,
                                                   const _Float16* __restrict__ vT,
                                                   float* __restrict__ out) {
  const int blk = blockIdx.x;                   // B * LQ/16 blocks
  const int b   = blk / (LQ / 16);
  const int q0  = (blk % (LQ / 16)) * 16;

  const _Float16* qbase = qh + ((size_t)b * LQ + q0) * DD;
  const _Float16* kbase = kh + (size_t)b * LK * DD;
  const _Float16* vbase = vT + (size_t)b * DD * LK;

  __shared__ __align__(16) _Float16 sQ[16 * DD];       // 16 KB
  __shared__ __align__(16) float    sS[16 * 128];      // 8 KB
  __shared__ __align__(16) _Float16 sP[2][16 * 128];   // 2 x 4 KB (dbl-buf)
  __shared__ float sM[16], sL[16], sAlpha[16];

  const int tid   = threadIdx.x;
  const int lane  = tid & 31;
  const int wave  = tid >> 5;
  const int lm    = lane & 15;
  const int lhalf = lane >> 4;

  // Stage q tile into LDS (row-major [m][d], 16B-aligned rows)
  for (int i = tid; i < 16 * DD; i += 256) sQ[i] = qbase[i];
  if (tid < 16) { sM[tid] = -1e30f; sL[tid] = 0.0f; }
  __syncthreads();

  v8f ctx[4] = {};
  const int dslice = wave * 64;

  for (int j0 = 0; j0 < LK; j0 += 128) {
    const bool has_next = (j0 + 128) < LK;
    const int  pb = (j0 >> 7) & 1;              // sP buffer for this tile

    // ---- score phase: S = q @ k^T for this wave's 16 keys
    v8f s = {};
    const int jn = j0 + wave * 16 + lm;         // B-frag column (key index)
    const _Float16* krow = kbase + (size_t)jn * DD;
    // branchless prefetch: distance 0 when on the last tile
    const _Float16* kpre = krow + (has_next ? (size_t)128 * DD : (size_t)0);
#pragma unroll 4
    for (int d0 = 0; d0 < DD; d0 += 32) {
      v16h a  = ld_afrag(&sQ[lm * DD + d0], lhalf * 16);
      v16h bf = ld_bfrag(krow + d0 + lhalf * 16);
      __builtin_prefetch(kpre + d0, 0, 3);      // next K tile (L2-resident)
      s = __builtin_amdgcn_wmma_f32_16x16x32_f16(
          false, a, false, bf, (short)0, s, false, false);
    }
#pragma unroll
    for (int r = 0; r < 8; ++r)
      sS[(r + lhalf * 8) * 128 + wave * 16 + lm] = s[r];
    __syncthreads();

    // ---- online softmax: wave w owns rows 2w and 2w+1
#pragma unroll
    for (int rr = 0; rr < 2; ++rr) {
      const int row = wave * 2 + rr;
      float vals[4];
      float mx = -1e30f;
#pragma unroll
      for (int i = 0; i < 4; ++i) {
        vals[i] = sS[row * 128 + lane + 32 * i];
        mx = fmaxf(mx, vals[i]);
      }
#pragma unroll
      for (int off = 16; off; off >>= 1) mx = fmaxf(mx, __shfl_xor(mx, off, 32));
      const float mold = sM[row];
      const float mnew = fmaxf(mold, mx);
      float lsum = 0.0f;
#pragma unroll
      for (int i = 0; i < 4; ++i) {
        float p = __expf(vals[i] - mnew);
        lsum += p;
        sP[pb][row * 128 + lane + 32 * i] = (_Float16)p;
      }
#pragma unroll
      for (int off = 16; off; off >>= 1) lsum += __shfl_xor(lsum, off, 32);
      if (lane == 0) {
        float alpha = __expf(mold - mnew);
        sL[row]     = sL[row] * alpha + lsum;
        sM[row]     = mnew;
        sAlpha[row] = alpha;
      }
    }
    __syncthreads();

    // ---- context phase: rescale accumulators, then ctx += P @ vT-slice
#pragma unroll
    for (int ct = 0; ct < 4; ++ct) {
#pragma unroll
      for (int r = 0; r < 8; ++r) ctx[ct][r] *= sAlpha[r + lhalf * 8];
    }
    const int vpre = has_next ? 128 : 0;        // branchless prefetch dist
#pragma unroll
    for (int s4 = 0; s4 < 4; ++s4) {            // K steps over 128 keys
      v16h a = ld_afrag(&sP[pb][lm * 128 + s4 * 32], lhalf * 16);
#pragma unroll
      for (int ct = 0; ct < 4; ++ct) {
        int d = dslice + ct * 16 + lm;
        const _Float16* vp =
            vbase + (size_t)d * LK + j0 + s4 * 32 + lhalf * 16;
        v16h bf = ld_bfrag(vp);
        __builtin_prefetch(vp + vpre, 0, 3);    // next V tile
        ctx[ct] = __builtin_amdgcn_wmma_f32_16x16x32_f16(
            false, a, false, bf, (short)0, ctx[ct], false, false);
      }
    }
    // no trailing barrier: sP is double-buffered, sS reuse is protected by
    // this tile's post-softmax barrier
  }

  // ---- epilogue: normalize and store f32 context
  float invL[8];
#pragma unroll
  for (int r = 0; r < 8; ++r) invL[r] = 1.0f / sL[r + lhalf * 8];
  float* obase = out + ((size_t)b * LQ + q0) * DD;
#pragma unroll
  for (int ct = 0; ct < 4; ++ct) {
    int d = dslice + ct * 16 + lm;
#pragma unroll
    for (int r = 0; r < 8; ++r) {
      int row = r + lhalf * 8;
      obase[(size_t)row * DD + d] = ctx[ct][r] * invL[r];
    }
  }
}

// ---------------------------------------------------------------------------
// Host launcher
// ---------------------------------------------------------------------------
extern "C" void kernel_launch(void* const* d_in, const int* in_sizes, int n_in,
                              void* d_out, int out_size, void* d_ws, size_t ws_size,
                              hipStream_t stream) {
  const float* query = (const float*)d_in[0];
  const float* keys  = (const float*)d_in[1];
  const float* Wq    = (const float*)d_in[2];
  const float* bq    = (const float*)d_in[3];
  const float* Wk    = (const float*)d_in[4];
  const float* bk    = (const float*)d_in[5];
  const float* Wv    = (const float*)d_in[6];
  const float* bv    = (const float*)d_in[7];
  float* out = (float*)d_out;

  // Workspace layout (all f16): WqT | WkT | WvT | qh | kh | vT
  _Float16* WqT = (_Float16*)d_ws;
  _Float16* WkT = WqT + (size_t)DD * DD;
  _Float16* WvT = WkT + (size_t)DD * DD;
  _Float16* qh  = WvT + (size_t)DD * DD;
  _Float16* kh  = qh + (size_t)BB * LQ * DD;
  _Float16* vTp = kh + (size_t)BB * LK * DD;

  const float scale = 0.044194173824159216f;    // 1/sqrt(512)

  const int wblocks = (DD * DD) / 256;
  wt_kernel<<<wblocks, 256, 0, stream>>>(Wq, WqT);
  wt_kernel<<<wblocks, 256, 0, stream>>>(Wk, WkT);
  wt_kernel<<<wblocks, 256, 0, stream>>>(Wv, WvT);

  // q = (query @ Wq + bq) * scale   (fold the 1/sqrt(d) score scaling into q)
  proj_kernel<float, false><<<(BB * LQ) / 16, 256, 0, stream>>>(query, WqT, bq, qh, scale);
  // k = keys @ Wk + bk
  proj_kernel<float, false><<<(BB * LK) / 16, 256, 0, stream>>>(keys, WkT, bk, kh, 1.0f);
  // v = k @ Wv + bv, stored transposed per batch: vT[b][d][j]
  proj_kernel<_Float16, true><<<(BB * LK) / 16, 256, 0, stream>>>(kh, WvT, bv, vTp, 1.0f);

  // flash attention: out[b][q][d]
  attn_kernel<<<(BB * LQ) / 16, 256, 0, stream>>>(qh, kh, vTp, out);
}
